// MyGRU_66546223284335
// MI455X (gfx1250) — matrix operand order repaired and checked by reading, not compiled, not run on hardware
//
#include <hip/hip_runtime.h>
#include <math.h>

// Problem dims (fixed by the reference)
#define TT 512
#define BB 64
#define II 1024
#define HH 1024
#define MTOT (TT * BB)   // 32768 rows for input projections

#define KSTEP 32
#define STR   40         // padded LDS row stride in f16 (80B: conflict-free frags)

typedef __attribute__((ext_vector_type(16))) _Float16 v16h;
typedef __attribute__((ext_vector_type(8)))  _Float16 v8h;
typedef __attribute__((ext_vector_type(8)))  float    v8f;

// ---------------------------------------------------------------------------
// CDNA5 async copy: global -> LDS (16B per lane), tracked by ASYNCcnt.
// VDST carries the LDS byte address (low 32 bits of the generic LDS pointer).
// ---------------------------------------------------------------------------
__device__ __forceinline__ void async_copy16(_Float16* lds_dst, const _Float16* g_src) {
  unsigned l = (unsigned)(uintptr_t)lds_dst;   // wave-relative LDS byte address
  asm volatile("global_load_async_to_lds_b128 %0, %1, off"
               :: "v"(l), "v"(g_src)
               : "memory");
}

__device__ __forceinline__ void wait_async0() {
  asm volatile("s_wait_asynccnt 0x0" ::: "memory");
}

// ---------------------------------------------------------------------------
// WMMA fragment loaders (CDNA5 16x16x32 f16, wave32; contiguous k = 16*hi + s for A and B)
// ---------------------------------------------------------------------------
__device__ __forceinline__ v16h load_a_frag(const _Float16* __restrict__ A, int lda,
                                            int row, int k0, int hi) {
  const _Float16* p = A + (size_t)row * lda + k0 + hi * 16;
  v8h a0 = *(const v8h*)p;          // K chunk 0 (16B)
  v8h a1 = *(const v8h*)(p + 8);    // K chunk 1 (16B)
  return __builtin_shufflevector(a0, a1, 0,1,2,3,4,5,6,7,8,9,10,11,12,13,14,15);
}

__device__ __forceinline__ v16h load_b_frag(const _Float16* __restrict__ W, int ldw,
                                            int col, int k0, int hi) {
  return *(const v16h*)(W + (size_t)col * ldw + k0 + hi * 16);
}

// LDS variants (tile row-major with padded stride STR)
__device__ __forceinline__ v16h lds_a_frag(const _Float16* sh, int lrow, int hi) {
  const _Float16* p = sh + lrow * STR + hi * 16;
  v8h a0 = *(const v8h*)p;
  v8h a1 = *(const v8h*)(p + 8);
  return __builtin_shufflevector(a0, a1, 0,1,2,3,4,5,6,7,8,9,10,11,12,13,14,15);
}

__device__ __forceinline__ v16h lds_b_frag(const _Float16* sh, int lrow, int hi) {
  const _Float16* p = sh + lrow * STR + hi * 16;
  v8h b0 = *(const v8h*)p;
  v8h b1 = *(const v8h*)(p + 8);
  return __builtin_shufflevector(b0, b1, 0,1,2,3,4,5,6,7,8,9,10,11,12,13,14,15);
}

__device__ __forceinline__ v8f wmma_f16(v16h a, v16h b, v8f c) {
  return __builtin_amdgcn_wmma_f32_16x16x32_f16(false, a, false, b, (short)0, c, false, false);
}

__device__ __forceinline__ float sigmoidf_(float x) {
  return 1.0f / (1.0f + __expf(-x));
}

// ---------------------------------------------------------------------------
// Elementwise helpers
// ---------------------------------------------------------------------------
__global__ __launch_bounds__(256) void k_cast_f16(const float* __restrict__ src,
                                                  _Float16* __restrict__ dst, int n) {
  int i = blockIdx.x * 256 + threadIdx.x;
  if (i < n) dst[i] = (_Float16)src[i];
}

__global__ __launch_bounds__(256) void k_init_h(const float* __restrict__ h0,
                                                float* __restrict__ h,
                                                _Float16* __restrict__ h16, int n) {
  int i = blockIdx.x * 256 + threadIdx.x;
  if (i < n) { float v = h0[i]; h[i] = v; h16[i] = (_Float16)v; }
}

__global__ __launch_bounds__(256) void k_copy_f32(const float* __restrict__ src,
                                                  float* __restrict__ dst, int n) {
  int i = blockIdx.x * 256 + threadIdx.x;
  if (i < n) dst[i] = src[i];
}

// ---------------------------------------------------------------------------
// Stage a 64-row x 32-f16 tile into LDS via async copies.
// 256 16B chunks, 128 threads -> 2 chunks each.
// ---------------------------------------------------------------------------
__device__ __forceinline__ void stage_tile(_Float16* shdst, const _Float16* gsrc,
                                           int ld, int k0, int tid) {
  #pragma unroll
  for (int p = 0; p < 2; ++p) {
    const int c    = tid + p * 128;
    const int row  = c >> 2;        // 0..63
    const int part = c & 3;         // 16B chunk within the 64B row slice
    async_copy16(shdst + row * STR + part * 8,
                 gsrc + (size_t)row * ld + k0 + part * 8);
  }
}

// ---------------------------------------------------------------------------
// Input-projection GEMM: Y[m,n] = sum_k X[m,k]*W[n,k] + bias[n], stored f16.
// Block = 128 threads (4 waves), tile 64(M) x 64(N).
// Double-buffered async global->LDS staging overlapped with WMMA.
// All fragment loads are hoisted ahead of the 4 WMMAs so the k-step needs a
// single dscnt drain and the WMMAs issue back-to-back on the matrix pipe.
// Grid: (N/64, M/64)
// ---------------------------------------------------------------------------
__global__ __launch_bounds__(128) void k_xproj(const _Float16* __restrict__ X,
                                               const _Float16* __restrict__ W,
                                               const float* __restrict__ bias,
                                               _Float16* __restrict__ Y,
                                               int M, int N, int K) {
  __shared__ _Float16 shA[2][64 * STR];
  __shared__ _Float16 shB[2][64 * STR];

  const int tid  = threadIdx.x;
  const int wave = tid >> 5;
  const int lane = tid & 31;
  const int r    = lane & 15;
  const int hi   = lane >> 4;
  const int n0   = blockIdx.x * 64;
  const int m0b  = blockIdx.y * 64;

  const _Float16* Ablk = X + (size_t)m0b * K;
  const _Float16* Bblk = W + (size_t)n0 * K;

  // Prologue: stage k-step 0 into buffer 0
  stage_tile(shA[0], Ablk, K, 0, tid);
  stage_tile(shB[0], Bblk, K, 0, tid);

  v8f acc0 = {}, acc1 = {}, acc2 = {}, acc3 = {};
  const int nk = K / KSTEP;
  for (int i = 0; i < nk; ++i) {
    const int cur = i & 1;
    const int nxt = cur ^ 1;
    wait_async0();        // my async writes into buffer `cur` are done
    __syncthreads();      // everyone's writes visible; prior reads of `nxt` done
    if (i + 1 < nk) {
      stage_tile(shA[nxt], Ablk, K, (i + 1) * KSTEP, tid);
      stage_tile(shB[nxt], Bblk, K, (i + 1) * KSTEP, tid);
    }
    // Hoist all LDS fragment loads, then issue the 4 WMMAs back-to-back.
    const v16h a  = lds_a_frag(shA[cur], wave * 16 + r, hi);
    const v16h b0 = lds_b_frag(shB[cur],  0 + r, hi);
    const v16h b1 = lds_b_frag(shB[cur], 16 + r, hi);
    const v16h b2 = lds_b_frag(shB[cur], 32 + r, hi);
    const v16h b3 = lds_b_frag(shB[cur], 48 + r, hi);
    acc0 = wmma_f16(a, b0, acc0);
    acc1 = wmma_f16(a, b1, acc1);
    acc2 = wmma_f16(a, b2, acc2);
    acc3 = wmma_f16(a, b3, acc3);
  }

  const int m0 = m0b + wave * 16;
  v8f accs[4] = { acc0, acc1, acc2, acc3 };
  #pragma unroll
  for (int j = 0; j < 4; ++j) {
    const int n = n0 + j * 16 + r;
    const float bn = bias[n];
    #pragma unroll
    for (int e = 0; e < 8; ++e) {
      const int m = m0 + hi * 8 + e;   // D layout: VGPR e -> M = e (+8 for hi lanes)
      Y[(size_t)m * N + n] = (_Float16)(accs[j][e] + bn);
    }
  }
}

// ---------------------------------------------------------------------------
// Recurrent step, part 1: z and r gates (weights stay hot in 192MB L2).
// ---------------------------------------------------------------------------
__global__ __launch_bounds__(128) void k_gates_zr(const _Float16* __restrict__ h16,
                                                  const _Float16* __restrict__ Wzh,
                                                  const _Float16* __restrict__ Wrh,
                                                  const _Float16* __restrict__ xz_t,
                                                  const _Float16* __restrict__ xr_t,
                                                  const float* __restrict__ bzh,
                                                  const float* __restrict__ brh,
                                                  const float* __restrict__ h,
                                                  float* __restrict__ zbuf,
                                                  _Float16* __restrict__ rh16) {
  const int wave = threadIdx.x >> 5;
  const int lane = threadIdx.x & 31;
  const int r  = lane & 15;
  const int hi = lane >> 4;
  const int m0 = wave * 16;
  const int n0 = blockIdx.x * 16;

  v8f accz = {}, accr = {};
  for (int k0 = 0; k0 < HH; k0 += KSTEP) {
    v16h a  = load_a_frag(h16, HH, m0 + r, k0, hi);
    v16h bz = load_b_frag(Wzh, HH, n0 + r, k0, hi);
    v16h br = load_b_frag(Wrh, HH, n0 + r, k0, hi);
    accz = wmma_f16(a, bz, accz);
    accr = wmma_f16(a, br, accr);
  }

  const int n = n0 + r;
  const float bzn = bzh[n];
  const float brn = brh[n];
  #pragma unroll
  for (int e = 0; e < 8; ++e) {
    const int m = m0 + hi * 8 + e;
    const size_t idx = (size_t)m * HH + n;
    const float z  = sigmoidf_((float)xz_t[idx] + accz[e] + bzn);
    const float rr = sigmoidf_((float)xr_t[idx] + accr[e] + brn);
    zbuf[idx] = z;
    rh16[idx] = (_Float16)(rr * h[idx]);
  }
}

// ---------------------------------------------------------------------------
// Recurrent step, part 2: candidate + hidden update.
// ---------------------------------------------------------------------------
__global__ __launch_bounds__(128) void k_gate_c(const _Float16* __restrict__ rh16,
                                                const _Float16* __restrict__ Wch,
                                                const _Float16* __restrict__ xc_t,
                                                const float* __restrict__ bch,
                                                const float* __restrict__ zbuf,
                                                float* __restrict__ h,
                                                _Float16* __restrict__ h16,
                                                float* __restrict__ out_t) {
  const int wave = threadIdx.x >> 5;
  const int lane = threadIdx.x & 31;
  const int r  = lane & 15;
  const int hi = lane >> 4;
  const int m0 = wave * 16;
  const int n0 = blockIdx.x * 16;

  v8f acc = {};
  for (int k0 = 0; k0 < HH; k0 += KSTEP) {
    v16h a = load_a_frag(rh16, HH, m0 + r, k0, hi);
    v16h b = load_b_frag(Wch, HH, n0 + r, k0, hi);
    acc = wmma_f16(a, b, acc);
  }

  const int n = n0 + r;
  const float bn = bch[n];
  #pragma unroll
  for (int e = 0; e < 8; ++e) {
    const int m = m0 + hi * 8 + e;
    const size_t idx = (size_t)m * HH + n;
    const float c  = tanhf((float)xc_t[idx] + acc[e] + bn);
    const float z  = zbuf[idx];
    const float hn = (1.0f - z) * h[idx] + z * c;
    h[idx]   = hn;
    h16[idx] = (_Float16)hn;
    out_t[idx] = hn;
  }
}

// ---------------------------------------------------------------------------
// Host-side launch
// ---------------------------------------------------------------------------
static inline size_t align256(size_t x) { return (x + 255) & ~(size_t)255; }

extern "C" void kernel_launch(void* const* d_in, const int* in_sizes, int n_in,
                              void* d_out, int out_size, void* d_ws, size_t ws_size,
                              hipStream_t stream) {
  (void)in_sizes; (void)n_in; (void)out_size; (void)ws_size;

  const float* x_f   = (const float*)d_in[0];
  const float* h0_f  = (const float*)d_in[1];
  const float* Wzi_f = (const float*)d_in[2];
  const float* bzi   = (const float*)d_in[3];
  const float* Wzh_f = (const float*)d_in[4];
  const float* bzh   = (const float*)d_in[5];
  const float* Wri_f = (const float*)d_in[6];
  const float* bri   = (const float*)d_in[7];
  const float* Wrh_f = (const float*)d_in[8];
  const float* brh   = (const float*)d_in[9];
  const float* Wci_f = (const float*)d_in[10];
  const float* bci   = (const float*)d_in[11];
  const float* Wch_f = (const float*)d_in[12];
  const float* bch   = (const float*)d_in[13];

  float* out = (float*)d_out;                   // [T,B,H] then [B,H]

  char* ws = (char*)d_ws;
  size_t off = 0;
  _Float16* x16   = (_Float16*)(ws + off); off = align256(off + (size_t)MTOT * II * 2);
  _Float16* Wzi16 = (_Float16*)(ws + off); off = align256(off + (size_t)HH * II * 2);
  _Float16* Wri16 = (_Float16*)(ws + off); off = align256(off + (size_t)HH * II * 2);
  _Float16* Wci16 = (_Float16*)(ws + off); off = align256(off + (size_t)HH * II * 2);
  _Float16* Wzh16 = (_Float16*)(ws + off); off = align256(off + (size_t)HH * HH * 2);
  _Float16* Wrh16 = (_Float16*)(ws + off); off = align256(off + (size_t)HH * HH * 2);
  _Float16* Wch16 = (_Float16*)(ws + off); off = align256(off + (size_t)HH * HH * 2);
  _Float16* xz16  = (_Float16*)(ws + off); off = align256(off + (size_t)MTOT * HH * 2);
  _Float16* xr16  = (_Float16*)(ws + off); off = align256(off + (size_t)MTOT * HH * 2);
  _Float16* xc16  = (_Float16*)(ws + off); off = align256(off + (size_t)MTOT * HH * 2);
  float*    h     = (float*)(ws + off);    off = align256(off + (size_t)BB * HH * 4);
  _Float16* h16   = (_Float16*)(ws + off); off = align256(off + (size_t)BB * HH * 2);
  _Float16* rh16  = (_Float16*)(ws + off); off = align256(off + (size_t)BB * HH * 2);
  float*    zbuf  = (float*)(ws + off);    off = align256(off + (size_t)BB * HH * 4);

  // ---- Phase 0: casts to f16 ----
  {
    const int nx = MTOT * II;
    k_cast_f16<<<(nx + 255) / 256, 256, 0, stream>>>(x_f, x16, nx);
    const int nw = HH * II;
    k_cast_f16<<<(nw + 255) / 256, 256, 0, stream>>>(Wzi_f, Wzi16, nw);
    k_cast_f16<<<(nw + 255) / 256, 256, 0, stream>>>(Wri_f, Wri16, nw);
    k_cast_f16<<<(nw + 255) / 256, 256, 0, stream>>>(Wci_f, Wci16, nw);
    k_cast_f16<<<(nw + 255) / 256, 256, 0, stream>>>(Wzh_f, Wzh16, nw);
    k_cast_f16<<<(nw + 255) / 256, 256, 0, stream>>>(Wrh_f, Wrh16, nw);
    k_cast_f16<<<(nw + 255) / 256, 256, 0, stream>>>(Wch_f, Wch16, nw);
    const int nh = BB * HH;
    k_init_h<<<(nh + 255) / 256, 256, 0, stream>>>(h0_f, h, h16, nh);
  }

  // ---- Phase 1: input projections (async-LDS pipelined WMMA GEMMs) ----
  {
    dim3 grid(HH / 64, MTOT / 64);
    k_xproj<<<grid, 128, 0, stream>>>(x16, Wzi16, bzi, xz16, MTOT, HH, II);
    k_xproj<<<grid, 128, 0, stream>>>(x16, Wri16, bri, xr16, MTOT, HH, II);
    k_xproj<<<grid, 128, 0, stream>>>(x16, Wci16, bci, xc16, MTOT, HH, II);
  }

  // ---- Phase 2: sequential scan ----
  const size_t step = (size_t)BB * HH;
  for (int t = 0; t < TT; ++t) {
    const _Float16* xz_t = xz16 + (size_t)t * step;
    const _Float16* xr_t = xr16 + (size_t)t * step;
    const _Float16* xc_t = xc16 + (size_t)t * step;
    k_gates_zr<<<HH / 16, 128, 0, stream>>>(h16, Wzh16, Wrh16, xz_t, xr_t,
                                            bzh, brh, h, zbuf, rh16);
    k_gate_c<<<HH / 16, 128, 0, stream>>>(rh16, Wch16, xc_t, bch, zbuf,
                                          h, h16, out + (size_t)t * step);
  }

  // ---- Final hidden state -> tail of d_out ----
  {
    const int nh = BB * HH;
    k_copy_f32<<<(nh + 255) / 256, 256, 0, stream>>>(h, out + (size_t)TT * step, nh);
  }
}
